// SwinTransformerBlock_2138893713593
// MI455X (gfx1250) — matrix operand order repaired and checked by reading, not compiled, not run on hardware
//
#include <hip/hip_runtime.h>
#include <hip/hip_bf16.h>

typedef __attribute__((ext_vector_type(16))) _Float16 v16h;
typedef __attribute__((ext_vector_type(8)))  _Float16 v8h;
typedef __attribute__((ext_vector_type(8)))  float    v8f;

#define TOKENS 200704      // B * H * W = 64*56*56
#define CCH    384
#define HID    1536
#define NHEAD  12
#define NWIN   4096        // B * 64 windows
#define WTOK   49          // 7*7

// GEMM tiling
#define BM 256
#define BN 128
#define BK 32
#define LDA 40             // A tile pitch (halves): 32 + 8 pad
#define LDB 136            // B tile pitch (halves): 128 + 8 pad (68 dwords -> conflict-free)

// ---------------------------------------------------------------------------
// WMMA helpers (CDNA5 wave32, 16x16x32 f16 -> f32)
// ---------------------------------------------------------------------------
__device__ __forceinline__ v8f wmma_f16(v16h a, v16h b, v8f c) {
  return __builtin_amdgcn_wmma_f32_16x16x32_f16(false, a, false, b, (short)0, c,
                                                false, false);
}

// A fragment (16x32): lane holds row m = lane&15; K halves {kb..kb+7, kb+16..kb+23},
// kb = 8*(lane>>4).  (ISA 7.12.2, 16-bit A-matrix table)
__device__ __forceinline__ v16h load_a_frag(const _Float16* s, int lda, int lane) {
  const int m  = lane & 15;
  const int kb = (lane >> 4) << 3;
  const _Float16* p = s + m * lda + kb;
  v16h r;
#pragma unroll
  for (int h = 0; h < 8; ++h) { r[h] = p[h]; r[h + 8] = p[h + 16]; }
  return r;
}

// B fragment (32x16): lane holds column n = lane&15; K {kb..kb+7, kb+16..kb+23},
// kb = 8*(lane>>4).
__device__ __forceinline__ v16h load_b_frag(const _Float16* s, int ld, int lane) {
  const int n  = lane & 15;
  const int kb = (lane >> 4) << 3;
  const _Float16* p = s + kb * ld + n;
  v16h r;
#pragma unroll
  for (int h = 0; h < 8; ++h) { r[h] = p[h * ld]; r[h + 8] = p[(h + 16) * ld]; }
  return r;
}

// Async 16B global -> LDS DMA (ASYNCcnt).  VDST = LDS byte address in a VGPR;
// generic pointers to __shared__ carry the LDS offset in their low 32 bits.
__device__ __forceinline__ void async_load_b128(const _Float16* lds_dst,
                                                const _Float16* gsrc) {
  const unsigned dst = (unsigned)(unsigned long long)(const void*)lds_dst;
  asm volatile("global_load_async_to_lds_b128 %0, %1, off"
               :
               : "v"(dst), "v"(gsrc)
               : "memory");
}
__device__ __forceinline__ void wait_async0() {
  asm volatile("s_wait_asynccnt 0x0" ::: "memory");
}

// ---------------------------------------------------------------------------
// f32 -> f16 weight convert
// ---------------------------------------------------------------------------
__global__ void cvt_kernel(const float* __restrict__ src, _Float16* __restrict__ dst,
                           int n) {
  int i = blockIdx.x * 256 + threadIdx.x;
  if (i < n) dst[i] = (_Float16)src[i];
}

// ---------------------------------------------------------------------------
// LayerNorm (one wave per token). REMAP=true also applies roll(-3,-3) +
// window partition: output row = window-token index.
// ---------------------------------------------------------------------------
template <bool REMAP>
__global__ __launch_bounds__(256) void ln_kernel(const float* __restrict__ x,
                                                 const float* __restrict__ sc,
                                                 const float* __restrict__ bs,
                                                 _Float16* __restrict__ out) {
  const int lane = threadIdx.x & 31;
  const int t = blockIdx.x * 8 + (threadIdx.x >> 5);
  const float* row = x + (size_t)t * CCH;
  float v[12];
  float s = 0.f;
#pragma unroll
  for (int j = 0; j < 12; ++j) { v[j] = row[lane + 32 * j]; s += v[j]; }
#pragma unroll
  for (int o = 16; o > 0; o >>= 1) s += __shfl_xor(s, o, 32);
  const float mu = s * (1.f / 384.f);
  float q = 0.f;
#pragma unroll
  for (int j = 0; j < 12; ++j) { float d = v[j] - mu; q += d * d; }
#pragma unroll
  for (int o = 16; o > 0; o >>= 1) q += __shfl_xor(q, o, 32);
  const float rs = rsqrtf(q * (1.f / 384.f) + 1e-6f);

  size_t ro;
  if (REMAP) {
    const int b = t / 3136, rem = t - b * 3136;
    const int h = rem / 56, w = rem - h * 56;
    const int hs = (h + 53) % 56, wsv = (w + 53) % 56;  // roll by -3
    ro = ((size_t)((b * 8 + hs / 7) * 8 + wsv / 7) * WTOK + (hs % 7) * 7 + (wsv % 7)) * CCH;
  } else {
    ro = (size_t)t * CCH;
  }
#pragma unroll
  for (int j = 0; j < 12; ++j) {
    const int c = lane + 32 * j;
    out[ro + c] = (_Float16)((v[j] - mu) * rs * sc[c] + bs[c]);
  }
}

// ---------------------------------------------------------------------------
// Tiled WMMA GEMM: C = A(f16, MxK) * W(f16, KxN) + bias.
// Block tile 256x128, BK=32, 8 waves (4m x 2n), 64x64 per wave (16 WMMAs/step).
// LDS ping-pong double buffering fed by async global->LDS DMA: the next
// K-slice streams into the other buffer while the current slice's 16 WMMAs
// run; completion via s_wait_asynccnt + barrier.  Epilogue MODE:
//   0: store f16            1: tanh-GELU, store f16
//   2: proj -> window-reverse + roll(+3,+3) scatter + residual, store f32
//   3: + residual, store f32
// ---------------------------------------------------------------------------
template <int MODE>
__global__ __launch_bounds__(256) void gemm_kernel(
    const _Float16* __restrict__ A, const _Float16* __restrict__ W,
    const float* __restrict__ bias, void* __restrict__ outp,
    const float* __restrict__ resid, int M, int N, int K) {
  __shared__ _Float16 sA[2][BM * LDA];   // 2 x 20 KB
  __shared__ _Float16 sB[2][BK * LDB];   // 2 x 8.5 KB
  const int tid = threadIdx.x;
  const int lane = tid & 31;
  const int wid = tid >> 5;
  const int wm = wid >> 1;  // 0..3 -> 64 rows each
  const int wn = wid & 1;   // 0..1 -> 64 cols each
  const int m0 = blockIdx.y * BM;
  const int n0 = blockIdx.x * BN;

  const v8f vz = {0.f, 0.f, 0.f, 0.f, 0.f, 0.f, 0.f, 0.f};
  v8f acc[4][4];
#pragma unroll
  for (int mt = 0; mt < 4; ++mt)
#pragma unroll
    for (int nt = 0; nt < 4; ++nt) acc[mt][nt] = vz;

  auto issue_tile = [&](int k0, int buf) {
#pragma unroll
    for (int it = 0; it < 4; ++it) {          // A tile 256x32 (64B/thread)
      const int idx = tid + it * 256;
      const int row = idx >> 2, seg = idx & 3;
      async_load_b128(&sA[buf][row * LDA + seg * 8],
                      A + (size_t)(m0 + row) * K + k0 + seg * 8);
    }
#pragma unroll
    for (int it = 0; it < 2; ++it) {          // B tile 32x128 (32B/thread)
      const int idx = tid + it * 256;
      const int row = idx >> 4, seg = idx & 15;
      async_load_b128(&sB[buf][row * LDB + seg * 8],
                      W + (size_t)(k0 + row) * N + n0 + seg * 8);
    }
  };
  auto compute = [&](int buf) {
    v16h bf[4];
#pragma unroll
    for (int nt = 0; nt < 4; ++nt)
      bf[nt] = load_b_frag(&sB[buf][wn * 64 + nt * 16], LDB, lane);
#pragma unroll
    for (int mt = 0; mt < 4; ++mt) {
      const v16h af = load_a_frag(&sA[buf][(wm * 64 + mt * 16) * LDA], LDA, lane);
#pragma unroll
      for (int nt = 0; nt < 4; ++nt) acc[mt][nt] = wmma_f16(af, bf[nt], acc[mt][nt]);
    }
  };

  issue_tile(0, 0);
  wait_async0();
  __syncthreads();
  int cur = 0;
  for (int k0 = BK; k0 < K; k0 += BK) {
    issue_tile(k0, cur ^ 1);   // DMA in flight during the 16 WMMAs below
    compute(cur);
    wait_async0();
    __syncthreads();
    cur ^= 1;
  }
  compute(cur);

#pragma unroll
  for (int mt = 0; mt < 4; ++mt) {
#pragma unroll
    for (int nt = 0; nt < 4; ++nt) {
      const int n = n0 + wn * 64 + nt * 16 + (lane & 15);
      const float bn = bias[n];
#pragma unroll
      for (int r = 0; r < 8; ++r) {
        const int m = m0 + wm * 64 + mt * 16 + r + ((lane >> 4) << 3);
        const float v = acc[mt][nt][r] + bn;
        if (MODE == 0) {
          ((_Float16*)outp)[(size_t)m * N + n] = (_Float16)v;
        } else if (MODE == 1) {
          const float g =
              0.5f * v * (1.f + tanhf(0.7978845608028654f * (v + 0.044715f * v * v * v)));
          ((_Float16*)outp)[(size_t)m * N + n] = (_Float16)g;
        } else if (MODE == 2) {
          const int win = m / WTOK, nn = m - win * WTOK;
          const int b = win >> 6, wim = win & 63;
          const int hs = (wim >> 3) * 7 + nn / 7;
          const int wsv = (wim & 7) * 7 + nn - (nn / 7) * 7;
          int hh = hs + 3; if (hh >= 56) hh -= 56;   // roll(+3,+3)
          int ww = wsv + 3; if (ww >= 56) ww -= 56;
          const size_t t = ((size_t)b * 3136 + hh * 56 + ww) * CCH + n;
          ((float*)outp)[t] = resid[t] + v;
        } else {
          const size_t t = (size_t)m * CCH + n;
          ((float*)outp)[t] = resid[t] + v;
        }
      }
    }
  }
}

// ---------------------------------------------------------------------------
// Attention: one block (4 waves) per (window, head). N=49 padded to 64.
// Wave w handles query rows [16w, 16w+16).
// ---------------------------------------------------------------------------
__global__ __launch_bounds__(128) void attn_kernel(const _Float16* __restrict__ qkv,
                                                   _Float16* __restrict__ o) {
  __shared__ _Float16 sQ[64 * 40];   // q rows (A layout source)
  __shared__ _Float16 sKT[32 * 72];  // k transposed: [d][n]  (B for q.k^T)
  __shared__ _Float16 sV[64 * 40];   // v rows: [n][d]        (B for p.v)
  __shared__ _Float16 sP[64 * 72];   // softmax probs
  __shared__ int sRid[64];

  const int tid = threadIdx.x, lane = tid & 31, wv = tid >> 5;
  const int win = blockIdx.x / NHEAD, nh = blockIdx.x % NHEAD;
  const _Float16* base = qkv + (size_t)win * WTOK * (3 * CCH) + nh * 32;

  for (int idx = tid; idx < 64 * 32; idx += 128) {
    const int n = idx >> 5, d = idx & 31;
    _Float16 qv = (_Float16)0.f, kv = qv, vv = qv;
    if (n < WTOK) {
      const _Float16* r = base + (size_t)n * (3 * CCH) + d;
      qv = r[0]; kv = r[CCH]; vv = r[2 * CCH];
    }
    sQ[n * 40 + d] = qv;
    sKT[d * 72 + n] = kv;
    sV[n * 40 + d] = vv;
  }
  if (tid < 64) {
    int rid = -1;
    if (tid < WTOK) {
      const int wim = win & 63;
      const int hh = (wim >> 3) * 7 + tid / 7;
      const int ww = (wim & 7) * 7 + tid % 7;
      const int rh = (hh < 49) ? 0 : (hh < 53 ? 1 : 2);
      const int rw = (ww < 49) ? 0 : (ww < 53 ? 1 : 2);
      rid = rh * 3 + rw;
    }
    sRid[tid] = rid;
  }
  __syncthreads();

  const v8f vz = {0.f, 0.f, 0.f, 0.f, 0.f, 0.f, 0.f, 0.f};
  const v16h aq = load_a_frag(sQ + wv * 16 * 40, 40, lane);
  v8f s[4];
#pragma unroll
  for (int nt = 0; nt < 4; ++nt) {
    const v16h bk = load_b_frag(sKT + nt * 16, 72, lane);
    s[nt] = wmma_f16(aq, bk, vz);
  }

  int jj[4], ridj[4];
#pragma unroll
  for (int nt = 0; nt < 4; ++nt) {
    jj[nt] = nt * 16 + (lane & 15);
    ridj[nt] = sRid[jj[nt]];
  }

  const float scale = 0.17677669529663687f;  // 32^-0.5
#pragma unroll
  for (int r = 0; r < 8; ++r) {
    const int i = wv * 16 + r + ((lane >> 4) << 3);
    const int ridi = sRid[i];
    float sv[4];
#pragma unroll
    for (int nt = 0; nt < 4; ++nt) {
      float xv = s[nt][r] * scale;
      xv += (jj[nt] < WTOK) ? ((ridj[nt] == ridi) ? 0.f : -100.f) : -1e30f;
      sv[nt] = xv;
    }
    float mx = fmaxf(fmaxf(sv[0], sv[1]), fmaxf(sv[2], sv[3]));
#pragma unroll
    for (int off = 8; off > 0; off >>= 1) mx = fmaxf(mx, __shfl_xor(mx, off, 32));
    float sum = 0.f;
#pragma unroll
    for (int nt = 0; nt < 4; ++nt) { sv[nt] = __expf(sv[nt] - mx); sum += sv[nt]; }
#pragma unroll
    for (int off = 8; off > 0; off >>= 1) sum += __shfl_xor(sum, off, 32);
    const float inv = 1.f / sum;
#pragma unroll
    for (int nt = 0; nt < 4; ++nt) sP[i * 72 + jj[nt]] = (_Float16)(sv[nt] * inv);
  }
  __syncthreads();

  const v16h ap0 = load_a_frag(sP + wv * 16 * 72, 72, lane);       // k = 0..31
  const v16h ap1 = load_a_frag(sP + wv * 16 * 72 + 32, 72, lane);  // k = 32..63
  v8f accO[2];
  accO[0] = vz; accO[1] = vz;
#pragma unroll
  for (int dt = 0; dt < 2; ++dt) {
    const v16h bv0 = load_b_frag(sV + dt * 16, 40, lane);
    const v16h bv1 = load_b_frag(sV + 32 * 40 + dt * 16, 40, lane);
    accO[dt] = wmma_f16(ap0, bv0, accO[dt]);
    accO[dt] = wmma_f16(ap1, bv1, accO[dt]);
  }
#pragma unroll
  for (int dt = 0; dt < 2; ++dt)
#pragma unroll
    for (int r = 0; r < 8; ++r) {
      const int i = wv * 16 + r + ((lane >> 4) << 3);
      if (i < WTOK) {
        const size_t off = ((size_t)win * WTOK + i) * CCH + nh * 32 + dt * 16 + (lane & 15);
        o[off] = (_Float16)accO[dt][r];
      }
    }
}

// ---------------------------------------------------------------------------
// Host side
// ---------------------------------------------------------------------------
extern "C" void kernel_launch(void* const* d_in, const int* in_sizes, int n_in,
                              void* d_out, int out_size, void* d_ws, size_t ws_size,
                              hipStream_t stream) {
  const float* x     = (const float*)d_in[0];
  const float* n1s   = (const float*)d_in[1];
  const float* n1b   = (const float*)d_in[2];
  const float* qkvw  = (const float*)d_in[3];
  const float* qkvb  = (const float*)d_in[4];
  const float* projw = (const float*)d_in[5];
  const float* projb = (const float*)d_in[6];
  const float* n2s   = (const float*)d_in[7];
  const float* n2b   = (const float*)d_in[8];
  const float* w1    = (const float*)d_in[9];
  const float* b1    = (const float*)d_in[10];
  const float* w2    = (const float*)d_in[11];
  const float* b2    = (const float*)d_in[12];
  float* out = (float*)d_out;

  // workspace layout (halves)
  _Float16* w_qkv  = (_Float16*)d_ws;                       // 384*1152
  _Float16* w_proj = w_qkv + 442368;                        // 384*384
  _Float16* w_mlp1 = w_proj + 147456;                       // 384*1536
  _Float16* w_mlp2 = w_mlp1 + 589824;                       // 1536*384
  _Float16* bufA   = w_mlp2 + 589824;                       // 200704*384 f16 (xw / ln2)
  _Float16* bufB   = bufA + (size_t)TOKENS * CCH;           // 200704*1152 f16 (qkv)
  float*    x2     = (float*)bufB;                          // aliases bufB after attn
  _Float16* bufC   = bufB + (size_t)TOKENS * (3 * CCH);     // 200704*1536 f16 (attn/hidden)

  cvt_kernel<<<(442368 + 255) / 256, 256, 0, stream>>>(qkvw, w_qkv, 442368);
  cvt_kernel<<<(147456 + 255) / 256, 256, 0, stream>>>(projw, w_proj, 147456);
  cvt_kernel<<<(589824 + 255) / 256, 256, 0, stream>>>(w1, w_mlp1, 589824);
  cvt_kernel<<<(589824 + 255) / 256, 256, 0, stream>>>(w2, w_mlp2, 589824);

  // LN1 + shift + window partition
  ln_kernel<true><<<TOKENS / 8, 256, 0, stream>>>(x, n1s, n1b, bufA);
  // qkv GEMM
  gemm_kernel<0><<<dim3(1152 / BN, TOKENS / BM), 256, 0, stream>>>(
      bufA, w_qkv, qkvb, bufB, nullptr, TOKENS, 1152, CCH);
  // attention
  attn_kernel<<<NWIN * NHEAD, 128, 0, stream>>>(bufB, bufC);
  // proj GEMM + reverse/roll/residual -> x2 (f32, overwrites qkv buffer)
  gemm_kernel<2><<<dim3(CCH / BN, TOKENS / BM), 256, 0, stream>>>(
      bufC, w_proj, projb, x2, x, TOKENS, CCH, CCH);
  // LN2
  ln_kernel<false><<<TOKENS / 8, 256, 0, stream>>>(x2, n2s, n2b, bufA);
  // MLP1 + GELU
  gemm_kernel<1><<<dim3(HID / BN, TOKENS / BM), 256, 0, stream>>>(
      bufA, w_mlp1, b1, bufC, nullptr, TOKENS, HID, CCH);
  // MLP2 + residual -> out
  gemm_kernel<3><<<dim3(CCH / BN, TOKENS / BM), 256, 0, stream>>>(
      bufC, w_mlp2, b2, out, x2, TOKENS, CCH, HID);
}
